// SimSiam_60224031424604
// MI455X (gfx1250) — matrix operand order repaired and checked by reading, not compiled, not run on hardware
//
#include <hip/hip_runtime.h>

// ---------------------------------------------------------------------------
// SimSiam pipeline for MI455X (gfx1250, wave32, WMMA).
// Pipeline:
//   1) rtot_kernel      : Rtot = (Rr*rw) @ (Rs*sw) @ (Rc*cw)       (3x3)
//   2) transform_kernel : x -> x_aug (einsum over 3-dim) -> bf16 feats [256][180000]
//   3) wconv_kernel     : W [180000][256] f32 -> Wt bf16 [256][180000] (transposed, LDS tiled)
//   4) gemm_kernel      : split-K bf16 WMMA GEMM -> 125 partial [256][256] f32 slices
//   5) reduce_kernel    : sum partials + bias -> z; copy z to d_out twice
//   6) loss_kernel      : SimSiam loss via rank-1 trick -> d_out[0]
// Workspace layout (bytes): ~217.4 MB total.
// ---------------------------------------------------------------------------

typedef __attribute__((ext_vector_type(16))) __bf16 v16bf;
typedef __attribute__((ext_vector_type(8)))  __bf16 v8bf;
typedef __attribute__((ext_vector_type(8)))  float  v8f;

union FragAB { v16bf v; v8bf h[2]; };
union FragC  { v8f   v; float f[8]; };

#define KDIM   180000
#define NROWS  256
#define SPLITS 125            // 125 * 45 k-steps * 32 = 180000
#define KSTEPS 45

__device__ __forceinline__ unsigned short f2bf(float f) {
    unsigned int u = __float_as_uint(f);
    unsigned int r = u + 0x7FFFu + ((u >> 16) & 1u);   // round-to-nearest-even
    return (unsigned short)(r >> 16);
}

// ---------------------------------------------------------------- 1) Rtot
__global__ void rtot_kernel(const float* __restrict__ rw, const float* __restrict__ sw,
                            const float* __restrict__ cw, const float* __restrict__ Rr,
                            const float* __restrict__ Rs, const float* __restrict__ Rc,
                            float* __restrict__ Rtot) {
    if (threadIdx.x != 0 || blockIdx.x != 0) return;
    float A[9], B[9], C[9], T1[9];
    for (int i = 0; i < 9; ++i) { A[i] = Rr[i] * rw[i]; B[i] = Rs[i] * sw[i]; C[i] = Rc[i] * cw[i]; }
    for (int i = 0; i < 3; ++i)
        for (int j = 0; j < 3; ++j) {
            float s = 0.f;
            for (int k = 0; k < 3; ++k) s += A[i*3+k] * B[k*3+j];
            T1[i*3+j] = s;
        }
    for (int i = 0; i < 3; ++i)
        for (int j = 0; j < 3; ++j) {
            float s = 0.f;
            for (int k = 0; k < 3; ++k) s += T1[i*3+k] * C[k*3+j];
            Rtot[i*3+j] = s;
        }
}

// ------------------------------------------- 2) transform x -> bf16 feats
// x logical shape (n, i, p, c=3, t, v, m); c-stride = 15000, (n,i,p)-stride = 45000.
// y_d = sum_c x_c * Rtot[c][d], stored at the same flat layout as bf16.
__global__ __launch_bounds__(256) void transform_kernel(const float* __restrict__ x,
                                                        const float* __restrict__ Rtot,
                                                        unsigned short* __restrict__ feats) {
    long gid = (long)blockIdx.x * 256 + threadIdx.x;
    if (gid >= 15360000L) return;                 // 256*4*15000 positions
    int pos = (int)(gid % 15000);
    int nip = (int)(gid / 15000);
    size_t base = (size_t)nip * 45000 + pos;
    float x0 = x[base], x1 = x[base + 15000], x2 = x[base + 30000];
    float r00 = Rtot[0], r01 = Rtot[1], r02 = Rtot[2];
    float r10 = Rtot[3], r11 = Rtot[4], r12 = Rtot[5];
    float r20 = Rtot[6], r21 = Rtot[7], r22 = Rtot[8];
    feats[base]           = f2bf(x0 * r00 + x1 * r10 + x2 * r20);
    feats[base + 15000]   = f2bf(x0 * r01 + x1 * r11 + x2 * r21);
    feats[base + 30000]   = f2bf(x0 * r02 + x1 * r12 + x2 * r22);
}

// --------------------------------- 3) W f32 [K][256] -> Wt bf16 [256][K]
__global__ __launch_bounds__(256) void wconv_kernel(const float* __restrict__ W,
                                                    unsigned short* __restrict__ Wt) {
    __shared__ unsigned short tile[32][33];       // +1 pad: no bank conflicts
    int bk = blockIdx.x % (KDIM / 32);            // 5625 k-tiles
    int bn = blockIdx.x / (KDIM / 32);            // 8 n-tiles
    int k0 = bk * 32, n0 = bn * 32;
    int nn = threadIdx.x & 31, kb = threadIdx.x >> 5;
    #pragma unroll
    for (int i = 0; i < 4; ++i) {
        int kk = kb + i * 8;
        tile[kk][nn] = f2bf(W[(size_t)(k0 + kk) * 256 + n0 + nn]);   // coalesced read
    }
    __syncthreads();
    int kk = threadIdx.x & 31, nb = threadIdx.x >> 5;
    #pragma unroll
    for (int i = 0; i < 4; ++i) {
        int n = nb + i * 8;
        Wt[(size_t)(n0 + n) * KDIM + k0 + kk] = tile[kk][n];         // coalesced write
    }
}

// ----------------------------------------------- 4) split-K bf16 WMMA GEMM
// grid.x = SPLITS*4 ; block = 256 (8 waves). Each block: 128x128 tile over a
// 1440-deep K slice. Wave grid 4(m) x 2(n); each wave: 2x4 16x16 acc tiles.
// Fragment loads are two global_load_b128 per operand tile, addressed per the
// CDNA5 16-bit WMMA VGPR layout (ISA 7.12.2).
__global__ __launch_bounds__(256) void gemm_kernel(const __bf16* __restrict__ A,   // feats [256][K]
                                                   const __bf16* __restrict__ B,   // Wt    [256][K]
                                                   float* __restrict__ partial) {  // [SPLITS][256][256]
    int bid = blockIdx.x;
    int s   = bid >> 2;
    int bm  = (bid >> 1) & 1;
    int bn  = bid & 1;
    int wave = threadIdx.x >> 5;
    int lane = threadIdx.x & 31;
    int wm = wave >> 1;            // 0..3 -> 32-row slab
    int wn = wave & 1;             // 0..1 -> 64-col slab
    int row0 = bm * 128 + wm * 32;
    int col0 = bn * 128 + wn * 64;
    int lr = lane & 15;
    int hi = lane >> 4;
    // A-matrix lane K offsets: lanes 0-15 -> {0,16}; lanes 16-31 -> {8,24}
    int aoff = hi ? 8 : 0;
    // B-matrix lane K offsets: lanes 0-15 -> {0,16}; lanes 16-31 -> {8,24}
    int boff = hi ? 8 : 0;

    const __bf16* Ar0 = A + (size_t)(row0 + lr) * KDIM;
    const __bf16* Ar1 = A + (size_t)(row0 + 16 + lr) * KDIM;
    const __bf16* Bc[4];
    #pragma unroll
    for (int tn = 0; tn < 4; ++tn) Bc[tn] = B + (size_t)(col0 + tn * 16 + lr) * KDIM;

    v8f acc[2][4];
    #pragma unroll
    for (int tm = 0; tm < 2; ++tm)
        #pragma unroll
        for (int tn = 0; tn < 4; ++tn) acc[tm][tn] = (v8f){};

    int kbase = s * KSTEPS * 32;
    for (int ks = 0; ks < KSTEPS; ++ks, kbase += 32) {
        FragAB a0, a1, b[4];
        a0.h[0] = *(const v8bf*)(Ar0 + kbase + aoff);
        a0.h[1] = *(const v8bf*)(Ar0 + kbase + aoff + 16);
        a1.h[0] = *(const v8bf*)(Ar1 + kbase + aoff);
        a1.h[1] = *(const v8bf*)(Ar1 + kbase + aoff + 16);
        #pragma unroll
        for (int tn = 0; tn < 4; ++tn) {
            b[tn].h[0] = *(const v8bf*)(Bc[tn] + kbase + boff);
            b[tn].h[1] = *(const v8bf*)(Bc[tn] + kbase + boff + 16);
        }
        #pragma unroll
        for (int tn = 0; tn < 4; ++tn)
            acc[0][tn] = __builtin_amdgcn_wmma_f32_16x16x32_bf16(
                false, a0.v, false, b[tn].v, (short)0, acc[0][tn], false, false);
        #pragma unroll
        for (int tn = 0; tn < 4; ++tn)
            acc[1][tn] = __builtin_amdgcn_wmma_f32_16x16x32_bf16(
                false, a1.v, false, b[tn].v, (short)0, acc[1][tn], false, false);
    }

    // C/D layout (ISA 7.12.2): VGPR r, lanes 0-15 -> M=r, N=lane; lanes 16-31 -> M=8+r.
    float* P = partial + (size_t)s * 65536;
    #pragma unroll
    for (int tm = 0; tm < 2; ++tm)
        #pragma unroll
        for (int tn = 0; tn < 4; ++tn) {
            FragC c; c.v = acc[tm][tn];
            int rbase = row0 + tm * 16 + hi * 8;
            int col = col0 + tn * 16 + lr;
            #pragma unroll
            for (int r = 0; r < 8; ++r)
                P[(size_t)(rbase + r) * 256 + col] = c.f[r];
        }
}

// ----------------------------------------------- 5) reduce partials + bias
__global__ __launch_bounds__(256) void reduce_kernel(const float* __restrict__ partial,
                                                     const float* __restrict__ bias,
                                                     float* __restrict__ z,
                                                     float* __restrict__ out) {
    int j = blockIdx.x * 256 + threadIdx.x;       // < 65536
    float s = bias[j & 255];
    for (int p = 0; p < SPLITS; ++p) s += partial[(size_t)p * 65536 + j];
    z[j] = s;
    out[1 + j] = s;                               // out tensor #1
    out[1 + 65536 + j] = s;                       // out tensor #2 (identical)
}

// ----------------------------------------------------------- 6) loss scalar
// q_i = z_i/||z_i||; sim_ij = q_i.q_j; diag_i = q_i.q_i; rowsum_i = q_i . (sum_j q_j)
// loss_i = diag_i - (rowsum_i - diag_i)/(N-1); output = 2 * mean(loss_i).
__global__ __launch_bounds__(256) void loss_kernel(const float* __restrict__ z,
                                                   float* __restrict__ out) {
    __shared__ float invn[256];
    __shared__ float sumq[256];
    __shared__ float red[256];
    int t = threadIdx.x;
    const float* zr = z + (size_t)t * 256;
    float s2 = 0.f;
    for (int j = 0; j < 256; ++j) { float v = zr[j]; s2 += v * v; }
    float in = rsqrtf(s2);
    invn[t] = in;
    __syncthreads();
    float cs = 0.f;
    for (int r = 0; r < 256; ++r) cs += z[(size_t)r * 256 + t] * invn[r];   // coalesced
    sumq[t] = cs;
    __syncthreads();
    float dot = 0.f;
    for (int j = 0; j < 256; ++j) dot += zr[j] * sumq[j];
    float rowsum = in * dot;
    float d = in * in * s2;                       // == 1 up to rounding
    red[t] = d - (rowsum - d) * (1.0f / 255.0f);
    __syncthreads();
    for (int off = 128; off; off >>= 1) {
        if (t < off) red[t] += red[t + off];
        __syncthreads();
    }
    if (t == 0) out[0] = 2.0f * red[0] * (1.0f / 256.0f);
}

// ---------------------------------------------------------------------------
extern "C" void kernel_launch(void* const* d_in, const int* in_sizes, int n_in,
                              void* d_out, int out_size, void* d_ws, size_t ws_size,
                              hipStream_t stream) {
    const float* x    = (const float*)d_in[0];
    const float* rotw = (const float*)d_in[1];
    const float* shw  = (const float*)d_in[2];
    const float* scw  = (const float*)d_in[3];
    const float* W    = (const float*)d_in[4];
    const float* bb   = (const float*)d_in[5];
    const float* Rr   = (const float*)d_in[6];
    const float* Rs   = (const float*)d_in[7];
    const float* Rc   = (const float*)d_in[8];
    float* out = (float*)d_out;

    constexpr size_t FEATS_BYTES = (size_t)NROWS * KDIM * 2;   // 92,160,000
    constexpr size_t OFF_RTOT  = 0;
    constexpr size_t OFF_FEATS = 256;
    constexpr size_t OFF_WT    = OFF_FEATS + FEATS_BYTES;
    constexpr size_t OFF_PART  = OFF_WT + FEATS_BYTES;
    constexpr size_t PART_BYTES = (size_t)SPLITS * 65536 * 4;  // 32,768,000
    constexpr size_t OFF_Z     = OFF_PART + PART_BYTES;

    char* ws = (char*)d_ws;
    float*          Rtot    = (float*)(ws + OFF_RTOT);
    unsigned short* feats   = (unsigned short*)(ws + OFF_FEATS);
    unsigned short* Wt      = (unsigned short*)(ws + OFF_WT);
    float*          partial = (float*)(ws + OFF_PART);
    float*          zbuf    = (float*)(ws + OFF_Z);

    rtot_kernel<<<1, 32, 0, stream>>>(rotw, shw, scw, Rr, Rs, Rc, Rtot);
    transform_kernel<<<60000, 256, 0, stream>>>(x, Rtot, feats);
    wconv_kernel<<<45000, 256, 0, stream>>>(W, Wt);
    gemm_kernel<<<SPLITS * 4, 256, 0, stream>>>((const __bf16*)feats, (const __bf16*)Wt, partial);
    reduce_kernel<<<256, 256, 0, stream>>>(partial, bb, zbuf, out);
    loss_kernel<<<1, 256, 0, stream>>>(zbuf, out);
}
